// MyLSTM_28449863369060
// MI455X (gfx1250) — matrix operand order repaired and checked by reading, not compiled, not run on hardware
//
#include <hip/hip_runtime.h>

// MI455X / gfx1250, wave32. gx GEMM via v_wmma_f32_16x16x32_bf16 with LDS
// staging (async-to-LDS for the bf16 weights), fused sigmoid/tanh epilogue
// and (b,t,h)->(t,b,h) transposed store.

typedef __bf16 bf16_t;
typedef __attribute__((ext_vector_type(16))) __bf16 v16bf;
typedef __attribute__((ext_vector_type(8)))  float  v8f;
typedef int v4i __attribute__((ext_vector_type(4)));

#define B_ 64
#define T_ 2048
#define I_ 256
#define H_ 256

#define MBLK 64          // bt rows per workgroup (grid = BT/64 = 2048 blocks)
#define KSTEP 32
#define LDS_STRIDE 40    // bf16 per LDS row: 32 data + 8 pad = 80 B
                         // -> 16B-aligned chunks, conflict-free 16-row sweeps

#if __has_builtin(__builtin_amdgcn_global_load_async_to_lds_b128) && \
    __has_builtin(__builtin_amdgcn_s_wait_asynccnt)
#define USE_ASYNC_LDS 1
// Builtin signature (from hipcc diagnostic): param0 = int4 AS1* (global src),
// param1 = int4 AS3* (LDS dst), then imm offset, imm cpol.
typedef __attribute__((address_space(1))) v4i* as1_v4i;
typedef __attribute__((address_space(3))) v4i* as3_v4i;
#endif

// ---- prep: convert W_fi (H x I, f32 row-major) to bf16 in workspace ----
__global__ __launch_bounds__(256) void wconv_kernel(const float* __restrict__ Wfi,
                                                    bf16_t* __restrict__ Wb) {
  int idx = blockIdx.x * 256 + threadIdx.x;
  Wb[idx] = (bf16_t)Wfi[idx];
}

// ---- prep: gadd[b][h] = b_fi[h] + b_fh[h] + dot(h0[b,:], W_fh[h,:]) ----
__global__ __launch_bounds__(256) void gadd_kernel(const float* __restrict__ h0,
                                                   const float* __restrict__ Wfh,
                                                   const float* __restrict__ bfi,
                                                   const float* __restrict__ bfh,
                                                   float* __restrict__ gadd) {
  int idx = blockIdx.x * 256 + threadIdx.x;   // 0 .. B*H-1
  int b = idx >> 8;
  int h = idx & (H_ - 1);
  const float* hr = h0  + (size_t)b * H_;
  const float* wr = Wfh + (size_t)h * H_;
  float acc = bfi[h] + bfh[h];
#pragma unroll 4
  for (int j = 0; j < H_; j += 4) {
    float4 hv = *(const float4*)(hr + j);
    float4 wv = *(const float4*)(wr + j);
    acc += hv.x * wv.x + hv.y * wv.y + hv.z * wv.z + hv.w * wv.w;
  }
  gadd[idx] = acc;
}

// ---- main kernel ----
// Block: 256 threads = 8 wave32. Block computes 64(bt) x 256(h) outputs.
// Wave w: row tile rt = w>>1 (16 rows), col half ch = w&1 (8 n-tiles of 16).
__global__ __launch_bounds__(256) void lstm_wmma_kernel(
    const float* __restrict__ x,      // (B,T,I) f32
    const float* __restrict__ c0,     // (B,H)
    const bf16_t* __restrict__ Wb,    // (H,I) bf16 row-major
    const float* __restrict__ gadd,   // (B,H)
    float* __restrict__ out) {        // (T,B,H) ++ h_last(B,H) ++ c_last(B,H)
  __shared__ bf16_t xs[MBLK * LDS_STRIDE];   // 5120 B: x tile, bf16
  __shared__ bf16_t wsl[H_ * LDS_STRIDE];    // 20480 B: W k-slab, bf16

  const int tid  = threadIdx.x;
  const int lane = tid & 31;
  const int wave = tid >> 5;
  const int rsel = lane & 15;
  const int hsel = lane >> 4;

  const int bt0 = blockIdx.x * MBLK;
  const int b   = bt0 >> 11;          // T_=2048; 64-row blocks never cross batch
  const int t0  = bt0 & (T_ - 1);

  const int rt = wave >> 1;           // row tile 0..3
  const int ch = wave & 1;            // col half 0..1

  // staging assignment: x -> thread copies one 8-f32 chunk (row tid>>2, chunk tid&3)
  const int xrow   = tid >> 2;
  const int xchunk = tid & 3;
  const float* xgp = x + (size_t)(bt0 + xrow) * I_ + xchunk * 8;
  bf16_t* xsp = &xs[xrow * LDS_STRIDE + xchunk * 8];
  // W -> thread copies row `tid` (4 x 16B chunks per k-step)
  const bf16_t* wgp = Wb + (size_t)tid * I_;
  bf16_t* wsp = &wsl[tid * LDS_STRIDE];

  v8f acc[8];
#pragma unroll
  for (int i = 0; i < 8; ++i) acc[i] = (v8f){};

  for (int k0 = 0; k0 < I_; k0 += KSTEP) {
    // ---- stage x tile: global f32 -> cvt -> LDS bf16 (one b128 store) ----
    float4 f0 = *(const float4*)(xgp + k0);
    float4 f1 = *(const float4*)(xgp + k0 + 4);
    union { bf16_t h[8]; uint4 u; } cvt;
    cvt.h[0] = (bf16_t)f0.x; cvt.h[1] = (bf16_t)f0.y;
    cvt.h[2] = (bf16_t)f0.z; cvt.h[3] = (bf16_t)f0.w;
    cvt.h[4] = (bf16_t)f1.x; cvt.h[5] = (bf16_t)f1.y;
    cvt.h[6] = (bf16_t)f1.z; cvt.h[7] = (bf16_t)f1.w;
    *(uint4*)xsp = cvt.u;

    // ---- stage W k-slab (bf16 byte copy): async-to-LDS ----
#ifdef USE_ASYNC_LDS
#pragma unroll
    for (int c = 0; c < 4; ++c) {
      // integer->pointer casts: AS1 = full 64-bit VA; AS3 = low 32 bits of the
      // generic LDS address == LDS byte offset (aperture rule).
      __builtin_amdgcn_global_load_async_to_lds_b128(
          (as1_v4i)(uintptr_t)(wgp + k0 + c * 8),
          (as3_v4i)(uintptr_t)(wsp + c * 8),
          0, 0);
    }
    __builtin_amdgcn_s_wait_asynccnt(0);
#else
    {
      uint4 w0 = *(const uint4*)(wgp + k0);
      uint4 w1 = *(const uint4*)(wgp + k0 + 8);
      uint4 w2 = *(const uint4*)(wgp + k0 + 16);
      uint4 w3 = *(const uint4*)(wgp + k0 + 24);
      *(uint4*)(wsp)      = w0;
      *(uint4*)(wsp + 8)  = w1;
      *(uint4*)(wsp + 16) = w2;
      *(uint4*)(wsp + 24) = w3;
    }
#endif
    __syncthreads();

    // ---- A fragment (16x32 bf16) from LDS, reused across 8 n-tiles ----
    // lane<16: row rsel, K {0..7, 16..23}; lane>=16: K {8..15, 24..31}
    v16bf afrag;
    {
      const bf16_t* ap = &xs[(rt * 16 + rsel) * LDS_STRIDE];
      union { uint4 u[2]; v16bf v; } au;
      au.u[0] = *(const uint4*)(ap + hsel * 8);
      au.u[1] = *(const uint4*)(ap + 16 + hsel * 8);
      afrag = au.v;
    }

    // ---- load ALL 8 B fragments first: batches the 16 ds_load_b128 so the
    // compiler issues staggered s_wait_dscnt instead of serializing each
    // WMMA behind its own LDS load (observed in round-3 asm) ----
    union bfr_t { uint4 u[2]; v16bf v; };
    bfr_t bu[8];
#pragma unroll
    for (int nt = 0; nt < 8; ++nt) {
      // B (32x16): lane<16: col rsel, K {0..7, 16..23}; lane>=16: K {8..15, 24..31}
      const bf16_t* bp = &wsl[(ch * 128 + nt * 16 + rsel) * LDS_STRIDE + hsel * 8];
      bu[nt].u[0] = *(const uint4*)(bp);
      bu[nt].u[1] = *(const uint4*)(bp + 16);
    }

    // ---- 8 WMMAs over this wave's 128-column half ----
#pragma unroll
    for (int nt = 0; nt < 8; ++nt) {
      acc[nt] = __builtin_amdgcn_wmma_f32_16x16x32_bf16(false, afrag, false, bu[nt].v,
                                                        (short)0, acc[nt], false, false);
    }
    __syncthreads();   // protect LDS before next k-step's staging
  }

  // ---- fused activation epilogue + transposed store ----
  const size_t hl_off = (size_t)T_ * B_ * H_;
  const size_t cl_off = hl_off + (size_t)B_ * H_;
  const int tbase = t0 + rt * 16;
#pragma unroll
  for (int nt = 0; nt < 8; ++nt) {
    const int h  = ch * 128 + nt * 16 + rsel;
    const float ga = gadd[b * H_ + h];
    const float cz = c0[b * H_ + h];
#pragma unroll
    for (int v = 0; v < 8; ++v) {
      const int t = tbase + v + hsel * 8;   // C/D: VGPR v -> row v + hsel*8
      float g  = acc[nt][v] + ga;
      float s  = 1.0f / (1.0f + __expf(-g));
      float m  = tanhf(g);
      float cc = s * (cz + m);
      float hh = s * tanhf(cc);
      out[((size_t)t * B_ + b) * H_ + h] = hh;
      if (t == T_ - 1) {
        out[hl_off + b * H_ + h] = hh;
        out[cl_off + b * H_ + h] = cc;
      }
    }
  }
}

extern "C" void kernel_launch(void* const* d_in, const int* in_sizes, int n_in,
                              void* d_out, int out_size, void* d_ws, size_t ws_size,
                              hipStream_t stream) {
  const float* x   = (const float*)d_in[0];
  const float* h0  = (const float*)d_in[1];
  const float* c0  = (const float*)d_in[2];
  const float* Wfi = (const float*)d_in[3];
  const float* bfi = (const float*)d_in[4];
  const float* Wfh = (const float*)d_in[5];
  const float* bfh = (const float*)d_in[6];
  float* out = (float*)d_out;

  // workspace: [0,128KB) = W_fi bf16; [128KB,192KB) = gadd f32 (B*H)
  bf16_t* Wb   = (bf16_t*)d_ws;
  float*  gadd = (float*)((char*)d_ws + (size_t)H_ * I_ * sizeof(bf16_t));

  wconv_kernel<<<(H_ * I_) / 256, 256, 0, stream>>>(Wfi, Wb);
  gadd_kernel<<<(B_ * H_) / 256, 256, 0, stream>>>(h0, Wfh, bfi, bfh, gadd);

  lstm_wmma_kernel<<<(B_ * T_) / MBLK, 256, 0, stream>>>(x, c0, Wb, gadd, out);
}
